// PWCDCNet_pose_12824772346460
// MI455X (gfx1250) — matrix operand order repaired and checked by reading, not compiled, not run on hardware
//
#include <hip/hip_runtime.h>
#include <hip/hip_bf16.h>
#include <stdint.h>

// ---------------------------------------------------------------------------
// PWC-Net (PWCDCNet_pose) forward for MI455X (gfx1250, wave32, WMMA).
// All 3x3 convs run as implicit-GEMM on v_wmma_f32_16x16x32_f16:
//   M = B*Ho*Wo output pixels, N = cout, K = cin*9 (per-tap K chunks of 32).
// Activations: padded NHWC f16 (16B-aligned b128 loads). Weights: f16
// [tap][coutPad16][Kpad32], zero-padded, staged per-tap in LDS (ds_load_b128).
// Border handling is branchless: invalid taps read from a zeroed scratch page
// via a per-lane pointer select, so the K-loop is straight-line and the
// scheduler can overlap next-chunk loads with the current WMMA.
// Input order assumed: d_in[0]=im1, d_in[1]=im2, then params in dict order
// (53 convs w/b, 6 deconvs w/b, fc1_w, fc1_b, fc2_w, fc2_b).
// ---------------------------------------------------------------------------

typedef _Float16 f16;
typedef _Float16 v16h __attribute__((ext_vector_type(16)));
typedef float    v8f  __attribute__((ext_vector_type(8)));

#define BB 8   // batch

static const int LH[6] = {128, 64, 32, 16, 8, 4};
static const int LW[6] = {416, 208, 104, 52, 26, 13};

// (cin, cout, stride, dilation) in CONV_SPECS order
static const int g_cs[53][4] = {
  {3,16,2,1},{16,16,1,1},{16,16,1,1},
  {16,32,2,1},{32,32,1,1},{32,32,1,1},
  {32,64,2,1},{64,64,1,1},{64,64,1,1},
  {64,96,2,1},{96,96,1,1},{96,96,1,1},
  {96,128,2,1},{128,128,1,1},{128,128,1,1},
  {81,128,1,1},{209,128,1,1},{337,96,1,1},{433,64,1,1},{497,32,1,1},
  {529,2,1,1},{529,128,1,1},
  {181,128,1,1},{309,128,1,1},{437,96,1,1},{533,64,1,1},{597,32,1,1},
  {629,2,1,1},{629,128,2,1},
  {149,128,1,1},{277,128,1,1},{405,96,1,1},{501,64,1,1},{565,32,1,1},
  {597,2,1,1},{597,128,2,1},{128,128,2,1},
  {117,128,1,1},{245,128,1,1},{373,96,1,1},{469,64,1,1},{533,32,1,1},
  {565,2,1,1},{565,128,2,1},{128,128,2,1},{128,128,2,1},
  {565,128,1,1},{128,128,1,2},{128,128,1,4},{128,96,1,8},{96,64,1,16},{64,32,1,1},{32,2,1,1}
};

// ---------------------------------------------------------------------------
// Kernels
// ---------------------------------------------------------------------------

// zero-fill (f16) -- used for the branchless border zero page
__global__ void k_zero_h(f16* __restrict__ p, int n) {
  int i = blockIdx.x * 256 + threadIdx.x;
  if (i < n) p[i] = (f16)0.f;
}

// fp32 OIHW weights -> f16 [9][coutPad][kpad], zero padded.
__global__ void k_convert_weights(const float* __restrict__ w, f16* __restrict__ o,
                                  int cout, int cin, int coutPad, int kpad) {
  int idx = blockIdx.x * 256 + threadIdx.x;
  int total = 9 * coutPad * kpad;
  if (idx >= total) return;
  int k = idx % kpad; int t = idx / kpad; int n = t % coutPad; int tap = t / coutPad;
  float v = 0.f;
  if (n < cout && k < cin) v = w[(size_t)(n * cin + k) * 9 + tap];
  o[idx] = (f16)v;
}

// NCHW f32 -> NHWC f16 (channel-strided), used for the input images.
__global__ void k_tonhwc(const float* __restrict__ src, int C, int H, int W,
                         f16* __restrict__ dst, int Cs, int Bn) {
  size_t idx = (size_t)blockIdx.x * 256 + threadIdx.x;
  size_t total = (size_t)Bn * C * H * W;
  if (idx >= total) return;
  int x = (int)(idx % W); size_t t = idx / W;
  int y = (int)(t % H);   t /= H;
  int c = (int)(t % C);   int b = (int)(t / C);
  dst[((size_t)(b * H + y) * W + x) * Cs + c] = (f16)src[idx];
}

// Copy a feature tensor into a concat buffer (both NCHW f32 and NHWC f16).
__global__ void k_inject(const float* __restrict__ src, int C, int H, int W,
                         float* __restrict__ dC, int Ct, int coff,
                         f16* __restrict__ dN, int CsN, int cbN, int Bn) {
  size_t idx = (size_t)blockIdx.x * 256 + threadIdx.x;
  size_t total = (size_t)Bn * C * H * W;
  if (idx >= total) return;
  int x = (int)(idx % W); size_t t = idx / W;
  int y = (int)(t % H);   t /= H;
  int c = (int)(t % C);   int b = (int)(t / C);
  float v = src[idx];
  dC[(((size_t)b * Ct + coff + c) * H + y) * W + x] = v;
  dN[((size_t)(b * H + y) * W + x) * CsN + cbN + c] = (f16)v;
}

// Implicit-GEMM 3x3 conv via WMMA. 8 waves/block, each 16x16 output tile.
__launch_bounds__(256)
__global__ void k_conv3x3_wmma(
    const f16* __restrict__ inN, int Hi, int Wi, int CsIn, int cbaseIn, int kpad,
    const f16* __restrict__ wt,        // f16 [9][coutPad][kpad]
    const f16* __restrict__ zeroPad,   // >= kpad zeros (branchless border reads)
    const float* __restrict__ bias, int cout, int coutPad,
    float* __restrict__ outC, int CoutTot, int coffOut,
    f16* __restrict__ outN, int CsOut, int cbaseOut,
    int Bn, int Ho, int Wo, int stride, int dil, int act)
{
  __shared__ f16 sW[16 * 640];           // weight tile for one tap (<=20KB)
  const int lane  = threadIdx.x & 31;
  const int wave  = threadIdx.x >> 5;
  const int n0    = blockIdx.x * 16;
  const int mBase = blockIdx.y * 128 + wave * 16;
  const int M     = Bn * Ho * Wo;

  // pixel carried by this lane's A-row (lanes 0-15 and 16-31 share rows)
  int mRow = mBase + (lane & 15);
  int mc = mRow < M ? mRow : M - 1;
  int b  = mc / (Ho * Wo);
  int r  = mc - b * (Ho * Wo);
  int oy = r / Wo, ox = r - oy * Wo;
  bool mOk = mRow < M;

  v8f acc = {};
  const int laneHi = lane >> 4;
  const int kA = laneHi ? 8  : 0;        // A layout: K {0..7,16..23} / {8..15,24..31}
  const int kB = laneHi ? 8  : 0;        // B layout: K {0..7,16..23} / {8..15,24..31}
  const int nLoc = lane & 15;
  const int kchunks = kpad >> 5;

  for (int tap = 0; tap < 9; ++tap) {
    // cooperatively stage this tap's 16 x kpad weight tile into LDS
    const f16* wsrc = wt + ((size_t)tap * coutPad + n0) * kpad;
    int nvec = (16 * kpad) >> 3;
    for (int v = threadIdx.x; v < nvec; v += 256) {
      int nn = v / (kpad >> 3);
      int kk = (v - nn * (kpad >> 3)) * 8;
      *(uint4*)&sW[nn * kpad + kk] = *(const uint4*)&wsrc[(size_t)nn * kpad + kk];
    }
    __syncthreads();

    int th = tap / 3, tw = tap - th * 3;
    int iy = oy * stride - dil + th * dil;
    int ix = ox * stride - dil + tw * dil;
    bool pv = mOk && iy >= 0 && iy < Hi && ix >= 0 && ix < Wi;
    // branchless: invalid taps read zeros from the zero page (same offsets)
    const f16* apReal = inN + ((size_t)(b * Hi + iy) * Wi + ix) * CsIn + cbaseIn;
    const f16* ap = pv ? apReal : zeroPad;

    for (int kc = 0; kc < kchunks; ++kc) {
      union { v16h h; uint4 q[2]; } Af, Bf;
      Af.q[0] = *(const uint4*)(ap + kc * 32 + kA);
      Af.q[1] = *(const uint4*)(ap + kc * 32 + kA + 16);
      const f16* bp = &sW[nLoc * kpad + kc * 32 + kB];
      Bf.q[0] = *(const uint4*)bp;
      Bf.q[1] = *(const uint4*)(bp + 16);
      acc = __builtin_amdgcn_wmma_f32_16x16x32_f16(
              false, Af.h, false, Bf.h, (short)0, acc, false, false);
    }
    __syncthreads();
  }

  int n = n0 + nLoc;
  if (n >= cout) return;
  float bv = bias ? bias[n] : 0.f;
  #pragma unroll
  for (int i = 0; i < 8; ++i) {
    int mm = mBase + i + laneHi * 8;     // C layout: VGPR i -> row i / i+8
    if (mm >= M) continue;
    int bb = mm / (Ho * Wo);
    int rr = mm - bb * (Ho * Wo);
    int yy = rr / Wo, xx = rr - yy * Wo;
    float v = acc[i] + bv;
    if (act) v = v > 0.f ? v : 0.1f * v;
    if (outC) outC[(((size_t)bb * CoutTot + coffOut + n) * Ho + yy) * Wo + xx] = v;
    if (outN) outN[((size_t)(bb * Ho + yy) * Wo + xx) * CsOut + cbaseOut + n] = (f16)v;
  }
}

// ConvTranspose2d(k=4, s=2, p=1). Direct (tiny FLOPs). Writes both formats.
__global__ void k_deconv4x4(const float* __restrict__ in, int CinTot, int cin,
                            int Hi, int Wi,
                            const float* __restrict__ w,  // (cin, cout, 4, 4)
                            const float* __restrict__ bias, int cout,
                            float* __restrict__ outC, int CoutTot, int coff,
                            f16* __restrict__ outN, int CsOut, int cbOut, int Bn)
{
  int Ho = Hi * 2, Wo = Wi * 2;
  int idx = blockIdx.x * 256 + threadIdx.x;
  int total = Bn * cout * Ho * Wo;
  if (idx >= total) return;
  int x = idx % Wo; int t = idx / Wo;
  int y = t % Ho;   t /= Ho;
  int co = t % cout; int b = t / cout;
  float s = bias[co];
  for (int kh = 0; kh < 4; ++kh) {
    int ty = y + 1 - kh; if (ty & 1) continue;
    int iy = ty >> 1;    if (iy < 0 || iy >= Hi) continue;
    for (int kw = 0; kw < 4; ++kw) {
      int tx = x + 1 - kw; if (tx & 1) continue;
      int ix = tx >> 1;    if (ix < 0 || ix >= Wi) continue;
      const float* ip = in + ((size_t)b * CinTot * Hi + iy) * Wi + ix;
      const float* wp = w + ((size_t)kh * 4 + kw) + (size_t)co * 16;
      for (int ci = 0; ci < cin; ++ci)
        s += ip[(size_t)ci * Hi * Wi] * wp[(size_t)ci * cout * 16];
    }
  }
  if (outC) outC[(((size_t)b * CoutTot + coff + co) * Ho + y) * Wo + x] = s;
  if (outN) outN[((size_t)(b * Ho + y) * Wo + x) * CsOut + cbOut + co] = (f16)s;
}

// Bilinear warp, torch grid_sample(align_corners=True, zeros) + validity mask.
__global__ void k_warp(const float* __restrict__ x, int C, int H, int W,
                       const float* __restrict__ dense, int Ct, int flowOff,
                       float scale, float* __restrict__ out, int Bn)
{
  size_t idx = (size_t)blockIdx.x * 256 + threadIdx.x;
  size_t total = (size_t)Bn * C * H * W;
  if (idx >= total) return;
  int xw = (int)(idx % W); size_t t = idx / W;
  int y  = (int)(t % H);   t /= H;
  int c  = (int)(t % C);   int b = (int)(t / C);
  size_t hw = (size_t)H * W;
  const float* fb = dense + ((size_t)b * Ct + flowOff) * hw;
  float u = fb[(size_t)y * W + xw] * scale;
  float v = fb[hw + (size_t)y * W + xw] * scale;
  float px = xw + u, py = y + v;
  float x0 = floorf(px), y0 = floorf(py);
  float wx = px - x0, wy = py - y0;
  const float* xb = x + ((size_t)b * C + c) * hw;
  float accv = 0.f, m = 0.f;
  #pragma unroll
  for (int dy = 0; dy < 2; ++dy)
    #pragma unroll
    for (int dx = 0; dx < 2; ++dx) {
      float xi = x0 + dx, yi = y0 + dy;
      float wgt = (dx ? wx : 1.f - wx) * (dy ? wy : 1.f - wy);
      bool valid = (xi >= 0.f) && (xi <= (float)(W - 1)) && (yi >= 0.f) && (yi <= (float)(H - 1));
      int xc = (int)xi; xc = xc < 0 ? 0 : (xc > W - 1 ? W - 1 : xc);
      int yc = (int)yi; yc = yc < 0 ? 0 : (yc > H - 1 ? H - 1 : yc);
      float val = valid ? xb[(size_t)yc * W + xc] : 0.f;
      accv += wgt * val;
      m += valid ? wgt : 0.f;
    }
  out[idx] = (m >= 0.9999f) ? accv : 0.f;
}

// 81-offset correlation (channel mean) + leaky-relu, into concat buffer.
__global__ void k_corr(const float* __restrict__ f1, const float* __restrict__ f2,
                       int C, int H, int W,
                       float* __restrict__ dC, int Ct, int coff,
                       f16* __restrict__ dN, int CsN, int cbN, int Bn)
{
  size_t idx = (size_t)blockIdx.x * 256 + threadIdx.x;
  size_t total = (size_t)Bn * 81 * H * W;
  if (idx >= total) return;
  int x = (int)(idx % W); size_t t = idx / W;
  int y = (int)(t % H);   t /= H;
  int o = (int)(t % 81);  int b = (int)(t / 81);
  int di = o / 9 - 4, dj = o % 9 - 4;
  int sy = y + di, sx = x + dj;
  float s = 0.f;
  if (sy >= 0 && sy < H && sx >= 0 && sx < W) {
    size_t hw = (size_t)H * W;
    const float* p1 = f1 + (size_t)b * C * hw + (size_t)y * W + x;
    const float* p2 = f2 + (size_t)b * C * hw + (size_t)sy * W + sx;
    for (int c = 0; c < C; ++c) s += p1[c * hw] * p2[c * hw];
  }
  s *= (1.f / (float)C);
  s = s > 0.f ? s : 0.1f * s;
  dC[(((size_t)b * Ct + coff + o) * H + y) * W + x] = s;
  dN[((size_t)(b * H + y) * W + x) * CsN + cbN + o] = (f16)s;
}

// out = a (+ b)
__global__ void k_addcopy(const float* __restrict__ a, const float* __restrict__ b,
                          float* __restrict__ o, int n) {
  int i = blockIdx.x * 256 + threadIdx.x;
  if (i >= n) return;
  o[i] = a[i] + (b ? b[i] : 0.f);
}

// fc1: out[b,o] = relu(pf[b,:] . w[o,:] + bias[o]); block per (o,b)
__global__ void k_fc1(const float* __restrict__ pf, const float* __restrict__ w,
                      const float* __restrict__ bias, float* __restrict__ out, int K) {
  int o = blockIdx.x, b = blockIdx.y;
  const float* pp = pf + (size_t)b * K;
  const float* wp = w + (size_t)o * K;
  float s = 0.f;
  for (int k = threadIdx.x; k < K; k += 256) s += pp[k] * wp[k];
  __shared__ float red[256];
  red[threadIdx.x] = s; __syncthreads();
  for (int st = 128; st > 0; st >>= 1) {
    if (threadIdx.x < st) red[threadIdx.x] += red[threadIdx.x + st];
    __syncthreads();
  }
  if (threadIdx.x == 0) {
    float v = red[0] + bias[o];
    out[(size_t)b * gridDim.x + o] = v > 0.f ? v : 0.f;
  }
}

// fc2: 8x6 outputs
__global__ void k_fc2(const float* __restrict__ h, const float* __restrict__ w,
                      const float* __restrict__ bias, float* __restrict__ out) {
  int i = threadIdx.x;
  if (i >= 48) return;
  int b = i / 6, j = i % 6;
  float s = bias[j];
  for (int k = 0; k < 128; ++k) s += h[b * 128 + k] * w[j * 128 + k];
  out[i] = s;
}

// ---------------------------------------------------------------------------
// Host orchestration
// ---------------------------------------------------------------------------

struct ConvW {
  const float* w; const float* b; f16* w16;
  int cin, cout, stride, dil, coutPad, kpad;
};

static inline int kpad32(int c) { return (c + 31) & ~31; }
static inline int cpad16(int c) { return (c + 15) & ~15; }

static const f16* g_zeroPad;   // set per launch before conv launches

static void launch_conv(hipStream_t st, const ConvW& c,
                        const f16* inN, int Hi, int Wi, int CsIn, int cbIn,
                        float* outC, int CoutTot, int coff,
                        f16* outN, int CsOut, int cbOut,
                        int Ho, int Wo, int act)
{
  int M = BB * Ho * Wo;
  dim3 g(c.coutPad / 16, (M + 127) / 128);
  k_conv3x3_wmma<<<g, 256, 0, st>>>(inN, Hi, Wi, CsIn, cbIn, c.kpad,
      c.w16, g_zeroPad, c.b, c.cout, c.coutPad,
      outC, CoutTot, coff, outN, CsOut, cbOut,
      BB, Ho, Wo, c.stride, c.dil, act);
}

extern "C" void kernel_launch(void* const* d_in, const int* in_sizes, int n_in,
                              void* d_out, int out_size, void* d_ws, size_t ws_size,
                              hipStream_t stream)
{
  (void)in_sizes; (void)n_in; (void)out_size; (void)ws_size;
  const float* im1 = (const float*)d_in[0];
  const float* im2 = (const float*)d_in[1];
  auto cwp = [&](int i) { return (const float*)d_in[2 + 2 * i]; };
  auto cbp = [&](int i) { return (const float*)d_in[3 + 2 * i]; };
  auto dwp = [&](int j) { return (const float*)d_in[108 + 2 * j]; };
  auto dbp = [&](int j) { return (const float*)d_in[109 + 2 * j]; };
  const float* fc1w = (const float*)d_in[120];
  const float* fc1b = (const float*)d_in[121];
  const float* fc2w = (const float*)d_in[122];
  const float* fc2b = (const float*)d_in[123];
  float* out = (float*)d_out;

  // deterministic bump allocator over d_ws
  char* base = (char*)d_ws; size_t off = 0;
  auto alloc  = [&](size_t bytes) -> void* {
    off = (off + 255) & ~(size_t)255; void* p = base + off; off += bytes; return p;
  };
  auto allocF = [&](size_t n) { return (float*)alloc(n * 4); };
  auto allocH = [&](size_t n) { return (f16*)alloc(n * 2); };

  // zero page for branchless border reads (>= max kpad halfs)
  f16* zp = allocH(2048);
  k_zero_h<<<(2048 + 255) / 256, 256, 0, stream>>>(zp, 2048);
  g_zeroPad = zp;

  // --- weight conversion ---------------------------------------------------
  ConvW cv[53];
  for (int i = 0; i < 53; ++i) {
    cv[i].w = cwp(i); cv[i].b = cbp(i);
    cv[i].cin = g_cs[i][0]; cv[i].cout = g_cs[i][1];
    cv[i].stride = g_cs[i][2]; cv[i].dil = g_cs[i][3];
    cv[i].kpad = kpad32(cv[i].cin); cv[i].coutPad = cpad16(cv[i].cout);
    cv[i].w16 = allocH((size_t)9 * cv[i].coutPad * cv[i].kpad);
  }
  for (int i = 0; i < 53; ++i) {
    int total = 9 * cv[i].coutPad * cv[i].kpad;
    k_convert_weights<<<(total + 255) / 256, 256, 0, stream>>>(
        cv[i].w, cv[i].w16, cv[i].cout, cv[i].cin, cv[i].coutPad, cv[i].kpad);
  }

  // --- workspace buffers ---------------------------------------------------
  f16* encA = allocH((size_t)BB * 128 * 416 * 32);  // holds image NHWC + ping
  f16* encB = allocH((size_t)BB * 64 * 208 * 32);   // pong
  float *c2f[2], *c3f[2], *c4f[2], *c5f[2];
  for (int i = 0; i < 2; ++i) {
    c2f[i] = allocF((size_t)BB * 32 * 32 * 104);
    c3f[i] = allocF((size_t)BB * 64 * 16 * 52);
    c4f[i] = allocF((size_t)BB * 96 * 8 * 26);
    c5f[i] = allocF((size_t)BB * 128 * 4 * 13);
  }
  float* d5C = allocF((size_t)BB * 529 * 4 * 13);   f16* d5N = allocH((size_t)BB * 4 * 13 * 640);
  float* d4C = allocF((size_t)BB * 629 * 8 * 26);   f16* d4N = allocH((size_t)BB * 8 * 26 * 640);
  float* d3C = allocF((size_t)BB * 597 * 16 * 52);  f16* d3N = allocH((size_t)BB * 16 * 52 * 640);
  float* d2C = allocF((size_t)BB * 565 * 32 * 104); f16* d2N = allocH((size_t)BB * 32 * 104 * 640);
  float* flow5  = allocF(832);
  float* flow4  = allocF(3328);
  float* flow3  = allocF(13312);
  float* flow2a = allocF(53248);
  float* dc7o   = allocF(53248);
  float* warpT  = allocF((size_t)BB * 32 * 32 * 104);   // max warp target
  float* poseBuf = allocF((size_t)BB * 512 * 52);
  float* ptC1 = allocF((size_t)BB * 128 * 16 * 52); f16* ptN1 = allocH((size_t)BB * 16 * 52 * 128);
  float* ptC2 = allocF((size_t)BB * 128 * 8 * 26);  f16* ptN2 = allocH((size_t)BB * 8 * 26 * 128);
  float* dcC1 = allocF((size_t)BB * 128 * 32 * 104); f16* dcN1 = allocH((size_t)BB * 32 * 104 * 128);
  float* dcC2 = allocF((size_t)BB * 128 * 32 * 104); f16* dcN2 = allocH((size_t)BB * 32 * 104 * 128);
  float* hid = allocF((size_t)BB * 128);

  // --- feature pyramid encoder --------------------------------------------
  auto run_enc = [&](const float* img, float* c2o, float* c3o, float* c4o, float* c5o) {
    {
      size_t total = (size_t)BB * 3 * 128 * 416;
      k_tonhwc<<<(int)((total + 255) / 256), 256, 0, stream>>>(img, 3, 128, 416, encA, 32, BB);
    }
    f16* cur = encA; f16* nxt = encB;
    for (int lvl = 0; lvl < 5; ++lvl) {
      for (int j = 0; j < 3; ++j) {
        int ci = lvl * 3 + j;
        int Hi = (j == 0) ? LH[lvl] : LH[lvl + 1];
        int Wi = (j == 0) ? LW[lvl] : LW[lvl + 1];
        int Ho = LH[lvl + 1], Wo = LW[lvl + 1];
        int CsIn = cv[ci].kpad;
        int CsOut = (ci < 14) ? cv[ci + 1].kpad : 128;
        float* outC = nullptr;
        if (ci == 5) outC = c2o; else if (ci == 8) outC = c3o;
        else if (ci == 11) outC = c4o; else if (ci == 14) outC = c5o;
        f16* outN = (ci == 14) ? nullptr : nxt;
        launch_conv(stream, cv[ci], cur, Hi, Wi, CsIn, 0,
                    outC, cv[ci].cout, 0, outN, CsOut, 0, Ho, Wo, 1);
        if (outN) { f16* t = cur; cur = nxt; nxt = t; }
      }
    }
  };
  run_enc(im1, c2f[0], c3f[0], c4f[0], c5f[0]);
  run_enc(im2, c2f[1], c3f[1], c4f[1], c5f[1]);

  const int basesIn[5] = {448, 320, 192, 96, 32};

  // --- level 5 -------------------------------------------------------------
  {
    int H = 4, W = 13;
    size_t total = (size_t)BB * 81 * H * W;
    k_corr<<<(int)((total + 255) / 256), 256, 0, stream>>>(
        c5f[0], c5f[1], 128, H, W, d5C, 529, 448, d5N, 640, 448, BB);
    for (int k = 0; k < 5; ++k) {
      const ConvW& c = cv[15 + k];
      launch_conv(stream, c, d5N, H, W, 640, basesIn[k],
                  d5C, 529, basesIn[k] - c.cout, d5N, 640, basesIn[k] - c.cout, H, W, 1);
    }
    launch_conv(stream, cv[20], d5N, H, W, 640, 0, flow5, 2, 0, nullptr, 0, 0, H, W, 0);
    launch_conv(stream, cv[21], d5N, H, W, 640, 0, poseBuf, 512, 0, nullptr, 0, 0, 4, 13, 1);
  }

  // --- levels 4, 3, 2 ------------------------------------------------------
  struct LvlDesc {
    int H, W, Hp, Wp, Ct, featC, idx0, pfIdx, dcvIdx;
    float scale;
    float* dC; f16* dN; float* flowPrev; float* densePrevC; int CtPrev;
    const float* c1; const float* c2v; float* flowOut;
  };
  LvlDesc lv[3] = {
    {8, 26, 4, 13, 629, 96, 22, 27, 0, 1.25f, d4C, d4N, flow5, d5C, 529, c4f[0], c4f[1], flow4},
    {16, 52, 8, 26, 597, 64, 29, 34, 2, 2.5f,  d3C, d3N, flow4, d4C, 629, c3f[0], c3f[1], flow3},
    {32, 104, 16, 52, 565, 32, 37, 42, 4, 5.0f, d2C, d2N, flow3, d3C, 597, c2f[0], c2f[1], flow2a},
  };
  for (int L = 0; L < 3; ++L) {
    LvlDesc& d = lv[L];
    int flowOff = 529 + d.featC, upOff = flowOff + 2;
    {
      int total = BB * 2 * d.H * d.W;
      k_deconv4x4<<<(total + 255) / 256, 256, 0, stream>>>(
          d.flowPrev, 2, 2, d.Hp, d.Wp, dwp(d.dcvIdx), dbp(d.dcvIdx), 2,
          d.dC, d.Ct, flowOff, d.dN, 640, flowOff, BB);
      k_deconv4x4<<<(total + 255) / 256, 256, 0, stream>>>(
          d.densePrevC, d.CtPrev, d.CtPrev, d.Hp, d.Wp, dwp(d.dcvIdx + 1), dbp(d.dcvIdx + 1), 2,
          d.dC, d.Ct, upOff, d.dN, 640, upOff, BB);
    }
    {
      size_t total = (size_t)BB * d.featC * d.H * d.W;
      k_inject<<<(int)((total + 255) / 256), 256, 0, stream>>>(
          d.c1, d.featC, d.H, d.W, d.dC, d.Ct, 529, d.dN, 640, 529, BB);
      k_warp<<<(int)((total + 255) / 256), 256, 0, stream>>>(
          d.c2v, d.featC, d.H, d.W, d.dC, d.Ct, flowOff, d.scale, warpT, BB);
    }
    {
      size_t total = (size_t)BB * 81 * d.H * d.W;
      k_corr<<<(int)((total + 255) / 256), 256, 0, stream>>>(
          d.c1, warpT, d.featC, d.H, d.W, d.dC, d.Ct, 448, d.dN, 640, 448, BB);
    }
    for (int k = 0; k < 5; ++k) {
      const ConvW& c = cv[d.idx0 + k];
      launch_conv(stream, c, d.dN, d.H, d.W, 640, basesIn[k],
                  d.dC, d.Ct, basesIn[k] - c.cout, d.dN, 640, basesIn[k] - c.cout,
                  d.H, d.W, 1);
    }
    launch_conv(stream, cv[d.pfIdx], d.dN, d.H, d.W, 640, 0,
                d.flowOut, 2, 0, nullptr, 0, 0, d.H, d.W, 0);
  }

  // --- pose heads ----------------------------------------------------------
  launch_conv(stream, cv[28], d4N, 8, 26, 640, 0, poseBuf, 512, 128, nullptr, 0, 0, 4, 13, 1);
  launch_conv(stream, cv[35], d3N, 16, 52, 640, 0, ptC2, 128, 0, ptN2, 128, 0, 8, 26, 1);
  launch_conv(stream, cv[36], ptN2, 8, 26, 128, 0, poseBuf, 512, 256, nullptr, 0, 0, 4, 13, 1);
  launch_conv(stream, cv[43], d2N, 32, 104, 640, 0, ptC1, 128, 0, ptN1, 128, 0, 16, 52, 1);
  launch_conv(stream, cv[44], ptN1, 16, 52, 128, 0, ptC2, 128, 0, ptN2, 128, 0, 8, 26, 1);
  launch_conv(stream, cv[45], ptN2, 8, 26, 128, 0, poseBuf, 512, 384, nullptr, 0, 0, 4, 13, 1);

  // --- context (dc) network ------------------------------------------------
  launch_conv(stream, cv[46], d2N, 32, 104, 640, 0, dcC1, 128, 0, dcN1, 128, 0, 32, 104, 1);
  launch_conv(stream, cv[47], dcN1, 32, 104, 128, 0, dcC2, 128, 0, dcN2, 128, 0, 32, 104, 1);
  launch_conv(stream, cv[48], dcN2, 32, 104, 128, 0, dcC1, 128, 0, dcN1, 128, 0, 32, 104, 1);
  launch_conv(stream, cv[49], dcN1, 32, 104, 128, 0, dcC2, 96, 0, dcN2, 128, 0, 32, 104, 1);
  launch_conv(stream, cv[50], dcN2, 32, 104, 128, 0, dcC1, 64, 0, dcN1, 128, 0, 32, 104, 1);
  launch_conv(stream, cv[51], dcN1, 32, 104, 128, 0, dcC2, 32, 0, dcN2, 128, 0, 32, 104, 1);
  launch_conv(stream, cv[52], dcN2, 32, 104, 128, 0, dc7o, 2, 0, nullptr, 0, 0, 32, 104, 0);

  // --- outputs -------------------------------------------------------------
  k_addcopy<<<(53248 + 255) / 256, 256, 0, stream>>>(flow2a, dc7o, out, 53248);
  k_addcopy<<<(13312 + 255) / 256, 256, 0, stream>>>(flow3, nullptr, out + 53248, 13312);
  k_addcopy<<<(3328 + 255) / 256, 256, 0, stream>>>(flow4, nullptr, out + 66560, 3328);
  k_addcopy<<<(832 + 255) / 256, 256, 0, stream>>>(flow5, nullptr, out + 69888, 832);

  dim3 gfc(128, BB);
  k_fc1<<<gfc, 256, 0, stream>>>(poseBuf, fc1w, fc1b, hid, 512 * 52);
  k_fc2<<<1, 64, 0, stream>>>(hid, fc2w, fc2b, out + 70720);
}
